// RelativeAttention_45621142618791
// MI455X (gfx1250) — matrix operand order repaired and checked by reading, not compiled, not run on hardware
//
#include <hip/hip_runtime.h>
#include <hip/hip_bf16.h>

// ---------------------------------------------------------------------------
// RelativeAttention for MI455X (gfx1250): bf16 WMMA everywhere.
// B=8, L=S=1024, D=512, H=8, HD=64, MAX_POS=512, causal.
// ---------------------------------------------------------------------------

typedef __bf16 bf16;
typedef __attribute__((ext_vector_type(16))) __bf16 v16bf;
typedef __attribute__((ext_vector_type(8)))  float  v8f;
typedef int v4i __attribute__((ext_vector_type(4)));
typedef __attribute__((address_space(1))) v4i* gv4i_p;   // global int4*
typedef __attribute__((address_space(3))) v4i* lv4i_p;   // LDS int4*

#define BB 8
#define LL 1024
#define DD 512
#define HH 8
#define HD 64
#define RP 576   // padded rel index range (513 valid, rest zero)

#ifndef __has_builtin
#define __has_builtin(x) 0
#endif
#if __has_builtin(__builtin_amdgcn_global_load_async_to_lds_b128) && \
    __has_builtin(__builtin_amdgcn_s_wait_asynccnt)
#define USE_ASYNC_LDS 1
#else
#define USE_ASYNC_LDS 0
#endif

__device__ __forceinline__ bf16 f2bf(float f) {
  unsigned u = __builtin_bit_cast(unsigned, f);
  unsigned r = u + 0x7FFFu + ((u >> 16) & 1u);          // round-to-nearest-even
  unsigned short h = (unsigned short)(r >> 16);
  return __builtin_bit_cast(bf16, h);
}

__device__ __forceinline__ v8f v8f_zero() {
  v8f z = {0.f, 0.f, 0.f, 0.f, 0.f, 0.f, 0.f, 0.f};
  return z;
}

// Copy 16 bytes global -> LDS, async (ASYNCcnt path) when available.
__device__ __forceinline__ void cp16_g2l(const bf16* g, bf16* l) {
#if USE_ASYNC_LDS
  __builtin_amdgcn_global_load_async_to_lds_b128((gv4i_p)g, (lv4i_p)l, 0, 0);
#else
  *(int4*)l = *(const int4*)g;
#endif
}
__device__ __forceinline__ void async_wait0() {
#if USE_ASYNC_LDS
  __builtin_amdgcn_s_wait_asynccnt(0);
#endif
}

// A-matrix fragment, 16x32 bf16, src row-major [m][k], ld in elements.
// ISA layout: lanes 0-15 -> M=lane, K = {0..7}U{16..23}; lanes 16-31 same M,
// K = {8..15}U{24..31}.
__device__ __forceinline__ v16bf load_afrag_bf(const bf16* src, int ld, int lane) {
  int m = lane & 15, hi = lane >> 4;
  const bf16* p0 = src + m * ld + hi * 8;
  const bf16* p1 = p0 + 16;
  v16bf a;
#pragma unroll
  for (int i = 0; i < 8; i++) { a[i] = p0[i]; a[8 + i] = p1[i]; }
  return a;
}

// B-matrix fragment, 32x16 bf16 (K x N), src layout [n][k] row-major.
// ISA layout: lanes 0-15 -> N=lane, K = {0..7}U{16..23}; lanes 16-31 same N,
// K = {8..15}U{24..31}.
__device__ __forceinline__ v16bf load_bfrag_bf(const bf16* src, int ld, int lane) {
  int n = lane & 15, hi = lane >> 4;
  const bf16* p = src + n * ld + hi * 8;
  v16bf b;
#pragma unroll
  for (int i = 0; i < 8; i++) { b[i] = p[i]; b[8 + i] = p[16 + i]; }
  return b;
}

__device__ __forceinline__ v8f wmma_bf(v16bf a, v16bf b, v8f c) {
  return __builtin_amdgcn_wmma_f32_16x16x32_bf16(false, a, false, b, (short)0, c,
                                                 false, false);
}

// ---------------------------------------------------------------------------
// f32 -> bf16 convert
// ---------------------------------------------------------------------------
__global__ void cvt_bf16_kernel(const float* __restrict__ in, bf16* __restrict__ out,
                                int n) {
  int i = blockIdx.x * 256 + threadIdx.x;
  if (i < n) out[i] = f2bf(in[i]);
}

// Build E [RP][64] (rows >512 zero) and ET [64][RP] (transposed) in bf16.
__global__ void build_e_kernel(const float* __restrict__ pos, bf16* __restrict__ E,
                               bf16* __restrict__ ET) {
  int i = blockIdx.x * 256 + threadIdx.x;
  if (i >= RP * HD) return;
  int r = i / HD, d = i % HD;
  float v = (r <= 512) ? pos[r * HD + d] : 0.f;
  E[r * HD + d] = f2bf(v);
  ET[d * RP + r] = f2bf(v);
}

// ---------------------------------------------------------------------------
// Generic GEMM: C[M,N] = A[M,K](bf16) @ W[N,K]^T(bf16) + bias
// block = 128 threads (4 waves); block tile 64(M) x 64(N); wave owns 16 M rows.
// Shared B tile (64n x 32k) is double-buffered in LDS via async-to-LDS copies
// (ASYNCcnt) so the k+1 copy overlaps the k-step WMMAs; A-fragments are
// register-prefetched one step ahead. All 4 B-fragments are loaded from LDS
// before the 4 WMMAs so DScnt waits can be partial.
// mode 0: bf16 out, head layout  [(m/L)*H + n/64][m%L][n%64]
// mode 2: bf16 out, head-T layout [(m/L)*H + n/64][n%64][m%L]
// mode 3: f32 out, row-major [m][n]
// ---------------------------------------------------------------------------
__global__ __launch_bounds__(128) void gemm_bf16_kernel(
    const bf16* __restrict__ A, int lda, const bf16* __restrict__ W, int ldb,
    const float* __restrict__ bias, int M, int N, int K, void* __restrict__ outp,
    int mode) {
  __shared__ bf16 btile[2][64][32];   // [buf][n][k], 4 KB each

  int tid = threadIdx.x, wave = tid >> 5, lane = tid & 31;
  int m0 = blockIdx.x * 64 + wave * 16;
  int n0 = blockIdx.y * 64;

  // Each thread copies 2 x 16B of the 4 KB B tile.
  int c0 = tid * 2;
  int bn0 = c0 >> 2, bk0 = (c0 & 3) * 8;
  int c1 = c0 + 1;
  int bn1 = c1 >> 2, bk1 = (c1 & 3) * 8;

  auto issue_btile = [&](int k0, int buf) {
    cp16_g2l(W + (size_t)(n0 + bn0) * ldb + k0 + bk0, &btile[buf][bn0][bk0]);
    cp16_g2l(W + (size_t)(n0 + bn1) * ldb + k0 + bk1, &btile[buf][bn1][bk1]);
  };

  v8f acc[4];
#pragma unroll
  for (int t = 0; t < 4; t++) acc[t] = v8f_zero();

  int nsteps = K >> 5;
  issue_btile(0, 0);
  v16bf a_cur = load_afrag_bf(A + (size_t)m0 * lda, lda, lane);

  for (int s = 0; s < nsteps; s++) {
    int buf = s & 1;
    __syncthreads();      // prior readers of buf^1 are done
    async_wait0();        // this thread's step-s copies landed in LDS
    __syncthreads();      // everyone's step-s copies visible
    if (s + 1 < nsteps) issue_btile((s + 1) * 32, buf ^ 1);

    v16bf a_next = a_cur;
    if (s + 1 < nsteps)
      a_next = load_afrag_bf(A + (size_t)m0 * lda + (s + 1) * 32, lda, lane);

    // load all B fragments first -> partial s_wait_dscnt between WMMAs
    v16bf bfr[4];
#pragma unroll
    for (int t = 0; t < 4; t++)
      bfr[t] = load_bfrag_bf(&btile[buf][t * 16][0], 32, lane);
#pragma unroll
    for (int t = 0; t < 4; t++) acc[t] = wmma_bf(a_cur, bfr[t], acc[t]);
    a_cur = a_next;
  }

  int nl = lane & 15, hi = lane >> 4;
#pragma unroll
  for (int t = 0; t < 4; t++) {
#pragma unroll
    for (int g = 0; g < 8; g++) {
      int m = m0 + g + hi * 8;
      int n = n0 + t * 16 + nl;
      float v = acc[t][g] + (bias ? bias[n] : 0.f);
      if (mode == 0) {
        size_t idx = (((size_t)(m >> 10) * HH + (n >> 6)) * LL + (m & 1023)) * HD +
                     (n & 63);
        ((bf16*)outp)[idx] = f2bf(v);
      } else if (mode == 2) {
        size_t idx = (((size_t)(m >> 10) * HH + (n >> 6)) * HD + (n & 63)) * LL +
                     (m & 1023);
        ((bf16*)outp)[idx] = f2bf(v);
      } else {
        ((float*)outp)[(size_t)m * N + n] = v;
      }
    }
  }
}

// ---------------------------------------------------------------------------
// Fused causal relative attention.
// One WG (2 waves) per (b, h, 16-row l-tile). Two-pass softmax.
// Waves split the s-chunks (32 wide) by parity; rel-score comes from
// precomputed qa[b,h,l,r]; rel-value via w[l,r] scatter in LDS + WMMA w@E^T.
// ---------------------------------------------------------------------------
__global__ __launch_bounds__(64) void attn_kernel(
    const bf16* __restrict__ Qb, const bf16* __restrict__ Kb,
    const bf16* __restrict__ Vt, const float* __restrict__ qa,
    const bf16* __restrict__ ET, bf16* __restrict__ attnbuf) {
  __shared__ float wlds[16][RP];       // skewed probability matrix
  __shared__ float ocomb[16][HD];      // cross-wave O1 accumulator
  __shared__ float stats[2][16][2];    // per-wave (rowmax, rowsum)
  __shared__ bf16 pstage[2][16][32];   // P tile staging for A-fragments

  int tid = threadIdx.x, wave = tid >> 5, lane = tid & 31;
  int nl = lane & 15, hi = lane >> 4;
  int l0 = blockIdx.x * 16, h = blockIdx.y, b = blockIdx.z;

  for (int i = tid; i < 16 * RP; i += 64) ((float*)wlds)[i] = 0.f;
  for (int i = tid; i < 16 * HD; i += 64) ((float*)ocomb)[i] = 0.f;
  __syncthreads();

  const bf16* Qp = Qb + (((size_t)b * HH + h) * LL + l0) * HD;
  const bf16* Kp = Kb + ((size_t)b * HH + h) * LL * HD;
  const bf16* Vp = Vt + ((size_t)b * HH + h) * HD * LL;
  const float* Rp = qa + ((size_t)b * HH + h) * LL * RP;

  v16bf q0 = load_afrag_bf(Qp, HD, lane);        // K = d 0..31
  v16bf q1 = load_afrag_bf(Qp + 32, HD, lane);   // K = d 32..63

  float mr[8], zr[8];
#pragma unroll
  for (int g = 0; g < 8; g++) { mr[g] = -3e38f; zr[g] = 0.f; }

  int nch = (l0 + 15) / 32 + 1;

  // ---------------- pass A: row max + denominator ----------------
  for (int c = wave; c < nch; c += 2) {
    int s0 = c * 32;
    const bf16* kp = Kp + (size_t)s0 * HD;
    __builtin_prefetch((const void*)(kp + (size_t)64 * HD), 0, 1);
    // issue all fragment loads, then WMMAs (partial loadcnt waits)
    v16bf kb0 = load_bfrag_bf(kp, HD, lane);
    v16bf kb1 = load_bfrag_bf(kp + 32, HD, lane);
    v16bf kb2 = load_bfrag_bf(kp + 16 * HD, HD, lane);
    v16bf kb3 = load_bfrag_bf(kp + 16 * HD + 32, HD, lane);
    v8f sc0 = v8f_zero(), sc1 = v8f_zero();
    sc0 = wmma_bf(q0, kb0, sc0);
    sc0 = wmma_bf(q1, kb1, sc0);
    sc1 = wmma_bf(q0, kb2, sc1);
    sc1 = wmma_bf(q1, kb3, sc1);

#pragma unroll
    for (int g = 0; g < 8; g++) {
      int l = l0 + g + hi * 8;
      float v0 = -3e38f, v1 = -3e38f;
      int sA = s0 + nl;
      if (sA <= l) {
        int r = sA - l + 512; if (r < 0) r = 0;
        v0 = (sc0[g] + Rp[(size_t)l * RP + r]) * 0.125f;
      }
      int sB = s0 + 16 + nl;
      if (sB <= l) {
        int r = sB - l + 512; if (r < 0) r = 0;
        v1 = (sc1[g] + Rp[(size_t)l * RP + r]) * 0.125f;
      }
      float tmax = fmaxf(v0, v1);
#pragma unroll
      for (int off = 1; off < 16; off <<= 1)
        tmax = fmaxf(tmax, __shfl_xor(tmax, off, 32));
      float mnew = fmaxf(mr[g], tmax);
      bool valid = mnew > -1e37f;   // row has live columns in this chunk
      float rs = valid ? (__expf(v0 - mnew) + __expf(v1 - mnew)) : 0.f;
#pragma unroll
      for (int off = 1; off < 16; off <<= 1) rs += __shfl_xor(rs, off, 32);
      float corr = __expf(mr[g] - mnew);   // == 1 when both are -3e38
      zr[g] = zr[g] * corr + rs;
      mr[g] = mnew;
    }
  }

  if (nl == 0) {
#pragma unroll
    for (int g = 0; g < 8; g++) {
      stats[wave][g + hi * 8][0] = mr[g];
      stats[wave][g + hi * 8][1] = zr[g];
    }
  }
  __syncthreads();

  float mf[8], zi[8];
#pragma unroll
  for (int g = 0; g < 8; g++) {
    int row = g + hi * 8;
    float ma = stats[0][row][0], za = stats[0][row][1];
    float mb = stats[1][row][0], zb = stats[1][row][1];
    float Mx = fmaxf(ma, mb);
    float Z = za * __expf(ma - Mx) + zb * __expf(mb - Mx);
    mf[g] = Mx;
    zi[g] = (Z > 0.f) ? 1.f / Z : 0.f;
  }

  // ---------------- pass B: P@V and w-scatter ----------------
  v8f o1[4];
#pragma unroll
  for (int t = 0; t < 4; t++) o1[t] = v8f_zero();

  for (int c = wave; c < nch; c += 2) {
    int s0 = c * 32;
    const bf16* kp = Kp + (size_t)s0 * HD;
    v16bf kb0 = load_bfrag_bf(kp, HD, lane);
    v16bf kb1 = load_bfrag_bf(kp + 32, HD, lane);
    v16bf kb2 = load_bfrag_bf(kp + 16 * HD, HD, lane);
    v16bf kb3 = load_bfrag_bf(kp + 16 * HD + 32, HD, lane);
    // hoist V fragment loads: global latency hides behind exp/softmax VALU
    v16bf vb[4];
#pragma unroll
    for (int t = 0; t < 4; t++)
      vb[t] = load_bfrag_bf(Vp + (size_t)(t * 16) * LL + s0, LL, lane);

    v8f sc0 = v8f_zero(), sc1 = v8f_zero();
    sc0 = wmma_bf(q0, kb0, sc0);
    sc0 = wmma_bf(q1, kb1, sc0);
    sc1 = wmma_bf(q0, kb2, sc1);
    sc1 = wmma_bf(q1, kb3, sc1);

#pragma unroll
    for (int g = 0; g < 8; g++) {
      int row = g + hi * 8;
      int l = l0 + row;
      int sA = s0 + nl;
      float p0 = 0.f; int r0 = 0;
      if (sA <= l) {
        r0 = sA - l + 512; if (r0 < 0) r0 = 0;
        p0 = __expf((sc0[g] + Rp[(size_t)l * RP + r0]) * 0.125f - mf[g]) * zi[g];
      }
      pstage[wave][row][nl] = f2bf(p0);
      if (p0 > 0.f) atomicAdd(&wlds[row][r0], p0);   // ds_add_f32

      int sB = s0 + 16 + nl;
      float p1 = 0.f; int r1 = 0;
      if (sB <= l) {
        r1 = sB - l + 512; if (r1 < 0) r1 = 0;
        p1 = __expf((sc1[g] + Rp[(size_t)l * RP + r1]) * 0.125f - mf[g]) * zi[g];
      }
      pstage[wave][row][16 + nl] = f2bf(p1);
      if (p1 > 0.f) atomicAdd(&wlds[row][r1], p1);
    }

    // P (16x32) as A-fragment from LDS; V^T tiles as B-fragments.
    v16bf pf = load_afrag_bf(&pstage[wave][0][0], 32, lane);
#pragma unroll
    for (int t = 0; t < 4; t++) o1[t] = wmma_bf(pf, vb[t], o1[t]);
  }

#pragma unroll
  for (int t = 0; t < 4; t++)
#pragma unroll
    for (int g = 0; g < 8; g++)
      atomicAdd(&ocomb[g + hi * 8][t * 16 + nl], o1[t][g]);
  __syncthreads();   // w and O1 complete

  // ---------------- O2 = w @ E^T (wave splits d halves) ----------------
  v8f o2[2];
  o2[0] = v8f_zero(); o2[1] = v8f_zero();
  for (int k0 = 0; k0 < RP; k0 += 32) {
    v16bf af;
    int mrow = nl;
#pragma unroll
    for (int i = 0; i < 8; i++) {
      af[i]     = f2bf(wlds[mrow][k0 + hi * 8 + i]);
      af[8 + i] = f2bf(wlds[mrow][k0 + 16 + hi * 8 + i]);
    }
    // load both ET fragments before the two WMMAs
    v16bf bb0 = load_bfrag_bf(ET + (size_t)(wave * 32) * RP + k0, RP, lane);
    v16bf bb1 = load_bfrag_bf(ET + (size_t)(wave * 32 + 16) * RP + k0, RP, lane);
    o2[0] = wmma_bf(af, bb0, o2[0]);
    o2[1] = wmma_bf(af, bb1, o2[1]);
  }

#pragma unroll
  for (int t = 0; t < 2; t++)
#pragma unroll
    for (int g = 0; g < 8; g++) {
      int row = g + hi * 8;
      int l = l0 + row;
      int d = wave * 32 + t * 16 + nl;
      float v = ocomb[row][d] + o2[t][g];
      attnbuf[((size_t)b * LL + l) * DD + h * HD + d] = f2bf(v);
    }
}

// ---------------------------------------------------------------------------
// Host launcher
// ---------------------------------------------------------------------------
extern "C" void kernel_launch(void* const* d_in, const int* in_sizes, int n_in,
                              void* d_out, int out_size, void* d_ws, size_t ws_size,
                              hipStream_t stream) {
  (void)in_sizes; (void)n_in; (void)out_size; (void)ws_size;
  const float* query = (const float*)d_in[0];
  const float* key   = (const float*)d_in[1];
  const float* value = (const float*)d_in[2];
  const float* Wq = (const float*)d_in[4];
  const float* bq = (const float*)d_in[5];
  const float* Wk = (const float*)d_in[6];
  const float* bk = (const float*)d_in[7];
  const float* Wv = (const float*)d_in[8];
  const float* bv = (const float*)d_in[9];
  const float* pos = (const float*)d_in[10];
  const float* Wo = (const float*)d_in[11];
  const float* bo = (const float*)d_in[12];

  char* ws = (char*)d_ws;
  size_t off = 0;
  auto carve = [&](size_t bytes) {
    void* p = ws + off;
    off = (off + bytes + 255) & ~(size_t)255;
    return p;
  };
  const size_t nx = (size_t)BB * LL * DD;     // 4M elems
  const size_t nw = (size_t)DD * DD;          // 256K elems
  bf16* xq = (bf16*)carve(nx * 2);
  bf16* xk = (bf16*)carve(nx * 2);
  bf16* xv = (bf16*)carve(nx * 2);
  bf16* WqB = (bf16*)carve(nw * 2);
  bf16* WkB = (bf16*)carve(nw * 2);
  bf16* WvB = (bf16*)carve(nw * 2);
  bf16* WoB = (bf16*)carve(nw * 2);
  bf16* Qb = (bf16*)carve(nx * 2);
  bf16* Kb = (bf16*)carve(nx * 2);
  bf16* Vt = (bf16*)carve(nx * 2);
  bf16* E  = (bf16*)carve((size_t)RP * HD * 2);
  bf16* ET = (bf16*)carve((size_t)HD * RP * 2);
  float* qa = (float*)carve((size_t)BB * HH * LL * RP * 4);   // 151 MB
  bf16* attnbuf = (bf16*)carve(nx * 2);

  // 1) convert to bf16
  cvt_bf16_kernel<<<(int)((nx + 255) / 256), 256, 0, stream>>>(query, xq, (int)nx);
  cvt_bf16_kernel<<<(int)((nx + 255) / 256), 256, 0, stream>>>(key, xk, (int)nx);
  cvt_bf16_kernel<<<(int)((nx + 255) / 256), 256, 0, stream>>>(value, xv, (int)nx);
  cvt_bf16_kernel<<<(int)((nw + 255) / 256), 256, 0, stream>>>(Wq, WqB, (int)nw);
  cvt_bf16_kernel<<<(int)((nw + 255) / 256), 256, 0, stream>>>(Wk, WkB, (int)nw);
  cvt_bf16_kernel<<<(int)((nw + 255) / 256), 256, 0, stream>>>(Wv, WvB, (int)nw);
  cvt_bf16_kernel<<<(int)((nw + 255) / 256), 256, 0, stream>>>(Wo, WoB, (int)nw);
  build_e_kernel<<<(RP * HD + 255) / 256, 256, 0, stream>>>(pos, E, ET);

  // 2) QKV projections (M=8192, N=512, K=512)
  dim3 gp(BB * LL / 64, DD / 64);
  gemm_bf16_kernel<<<gp, 128, 0, stream>>>(xq, DD, WqB, DD, bq, BB * LL, DD, DD, Qb, 0);
  gemm_bf16_kernel<<<gp, 128, 0, stream>>>(xk, DD, WkB, DD, bk, BB * LL, DD, DD, Kb, 0);
  gemm_bf16_kernel<<<gp, 128, 0, stream>>>(xv, DD, WvB, DD, bv, BB * LL, DD, DD, Vt, 2);

  // 3) qa[b,h,l,r] = Q . E^T   (M=65536, N=576, K=64)
  dim3 gq(BB * HH * LL / 64, RP / 64);
  gemm_bf16_kernel<<<gq, 128, 0, stream>>>(Qb, HD, E, HD, nullptr,
                                           BB * HH * LL, RP, HD, qa, 3);

  // 4) fused attention
  dim3 ga(LL / 16, HH, BB);
  attn_kernel<<<ga, 64, 0, stream>>>(Qb, Kb, Vt, qa, ET, attnbuf);

  // 5) output projection -> f32 d_out
  gemm_bf16_kernel<<<gp, 128, 0, stream>>>(attnbuf, DD, WoB, DD, bo, BB * LL, DD, DD,
                                           d_out, 3);
}
